// TransformerFromScratch_51960514347781
// MI455X (gfx1250) — matrix operand order repaired and checked by reading, not compiled, not run on hardware
//
#include <hip/hip_runtime.h>
#include <math.h>

// ---------------------------------------------------------------------------
// Types for CDNA5 WMMA (wave32): v_wmma_f32_16x16x32_f16
// ---------------------------------------------------------------------------
typedef __attribute__((ext_vector_type(16))) _Float16 v16h;
typedef __attribute__((ext_vector_type(8)))  _Float16 v8h;
typedef __attribute__((ext_vector_type(8)))  float    v8f;
typedef __attribute__((ext_vector_type(4)))  float    f32x4;

#define BM 128
#define BN 64
#define BK 32
#define AP 40   // padded LDS row pitch (halves) for A tile: conflict-free frag loads
#define BP 48   // padded LDS row pitch (halves) for B tile

#define GF_TRANSB 1   // B is row-major [N][K] (i.e., compute A * B^T)
#define GF_RELU   2
#define GF_BIAS   4
#define GF_RESID  8

// ---------------------------------------------------------------------------
// General batched-strided GEMM:
//   C = alpha * (A @ op(B)) [+ bias[col]] [+ resid] [relu]
// A: [M,K] row-major (lda), B: [K,N] (ldb) or [N,K] if GF_TRANSB,
// C/resid: [M,N] (ldc). Batch z -> (z/heads, z%heads) with separate strides.
//
// 256 threads = 8 waves in a 4x2 grid; each wave owns a 32x32 register tile
// (4 f32 accumulators) and issues 4 v_wmma_f32_16x16x32_f16 per K-step from
// 2 A-fragments + 2 B-fragments (each reused twice). LDS is double-buffered
// and the next tile's global loads are issued before the WMMAs so VMEM
// latency hides under matrix work. Tile-fullness is block-uniform; only the
// generator's ragged N edge (N=10000) takes the guarded path.
// ---------------------------------------------------------------------------
__global__ __launch_bounds__(256) void wmma_gemm_kernel(
    const float* __restrict__ A, const float* __restrict__ B,
    const float* __restrict__ bias, const float* __restrict__ resid,
    float* __restrict__ C,
    int M, int N, int K, int lda, int ldb, int ldc,
    long long sAb, long long sAh, long long sBb, long long sBh,
    long long sCb, long long sCh, int heads, float alpha, int flags)
{
    __shared__ _Float16 As[2][BM * AP];   // [row][k] (padded pitch AP)
    __shared__ _Float16 Bs[2][BN * BP];   // [n][k]   (padded pitch BP)

    {
        const int z  = blockIdx.z;
        const int bo = z / heads;
        const int ho = z - bo * heads;
        A += bo * sAb + ho * sAh;
        B += bo * sBb + ho * sBh;
        C += bo * sCb + ho * sCh;
        if (resid) resid += bo * sCb + ho * sCh;
    }

    const int tile_n0 = blockIdx.x * BN;
    const int tile_m0 = blockIdx.y * BM;
    const int tid  = threadIdx.x;
    const int lane = tid & 31;
    const int wave = tid >> 5;
    const int wr   = wave >> 1;          // 0..3 -> row block of 32
    const int wc   = wave & 1;           // 0..1 -> col block of 32
    const int l15  = lane & 15;

    const bool transb = (flags & GF_TRANSB) != 0;
    const bool fullM  = (tile_m0 + BM) <= M;      // block-uniform
    const bool fullN  = (tile_n0 + BN) <= N;      // block-uniform

    // staging thread mapping
    const int r_a = tid >> 1;            // A: 0..127 rows
    const int c_a = (tid & 1) << 4;      // A: 0 or 16 (16 floats per thread)
    const int r_b = tid >> 2;            // NT-B: 0..63 rows(n)
    const int c_b = (tid & 3) << 3;      // NT-B: 0,8,16,24
    const int kk  = tid & 31;            // NN-B: k within tile
    const int n0  = (tid >> 5) << 3;     // NN-B: first of 8 n's

    v8f acc00 = {0.f,0.f,0.f,0.f,0.f,0.f,0.f,0.f};
    v8f acc01 = acc00, acc10 = acc00, acc11 = acc00;

    const int nk = (K + BK - 1) / BK;

    // ---- load phase: fetch one K-tile of A and B into float registers
    float fa[16], fb[8];
    auto load_tile = [&](int k0) {
        const bool kfull = (k0 + BK) <= K;
        // A: 16 contiguous floats per thread
        {
            const int gr = tile_m0 + r_a;
            const float* p = A + (long long)gr * lda + (k0 + c_a);
            if (fullM && kfull) {
                #pragma unroll
                for (int q = 0; q < 4; ++q) {
                    f32x4 x = *(const f32x4*)(p + 4 * q);
                    fa[4*q+0] = x[0]; fa[4*q+1] = x[1];
                    fa[4*q+2] = x[2]; fa[4*q+3] = x[3];
                }
            } else {
                #pragma unroll
                for (int j = 0; j < 16; ++j)
                    fa[j] = (gr < M && (k0 + c_a + j) < K) ? p[j] : 0.0f;
            }
        }
        // B
        if (transb) {
            const int gn = tile_n0 + r_b;
            const float* p = B + (long long)gn * ldb + (k0 + c_b);
            if (fullN && kfull) {
                f32x4 x = *(const f32x4*)p;
                f32x4 y = *(const f32x4*)(p + 4);
                fb[0] = x[0]; fb[1] = x[1]; fb[2] = x[2]; fb[3] = x[3];
                fb[4] = y[0]; fb[5] = y[1]; fb[6] = y[2]; fb[7] = y[3];
            } else {
                #pragma unroll
                for (int j = 0; j < 8; ++j)
                    fb[j] = (gn < N && (k0 + c_b + j) < K) ? p[j] : 0.0f;
            }
        } else {
            const float* p = B + (long long)(k0 + kk) * ldb + (tile_n0 + n0);
            if (fullN && kfull) {
                f32x4 x = *(const f32x4*)p;
                f32x4 y = *(const f32x4*)(p + 4);
                fb[0] = x[0]; fb[1] = x[1]; fb[2] = x[2]; fb[3] = x[3];
                fb[4] = y[0]; fb[5] = y[1]; fb[6] = y[2]; fb[7] = y[3];
            } else {
                const bool kok = (k0 + kk) < K;
                #pragma unroll
                for (int j = 0; j < 8; ++j)
                    fb[j] = (kok && (tile_n0 + n0 + j) < N) ? p[j] : 0.0f;
            }
        }
    };

    // ---- store phase: convert to f16 and commit to LDS buffer `b`
    auto store_tile = [&](int b) {
        v8h h0, h1;
        #pragma unroll
        for (int j = 0; j < 8; ++j) { h0[j] = (_Float16)fa[j]; h1[j] = (_Float16)fa[8+j]; }
        *(v8h*)&As[b][r_a * AP + c_a]     = h0;
        *(v8h*)&As[b][r_a * AP + c_a + 8] = h1;
        if (transb) {
            v8h hb;
            #pragma unroll
            for (int j = 0; j < 8; ++j) hb[j] = (_Float16)fb[j];
            *(v8h*)&Bs[b][r_b * BP + c_b] = hb;
        } else {
            #pragma unroll
            for (int j = 0; j < 8; ++j)
                Bs[b][(n0 + j) * BP + kk] = (_Float16)fb[j];   // LDS transpose
        }
    };

    // prologue: stage tile 0
    load_tile(0);
    store_tile(0);
    __syncthreads();

    const int r0 = wr << 5;              // wave row offset (0,32,64,96)
    const int c0 = wc << 5;              // wave col offset (0,32)
    int buf = 0;
    for (int t = 0; t < nk; ++t) {
        // issue next tile's global loads before doing matrix work
        const bool have_next = (t + 1) < nk;
        if (have_next) load_tile((t + 1) * BK);

        // ---- A fragments (16x32 f16): lanes 0-15 row M=l, K 0..15;
        //      lanes 16-31 row M=l-16, K 16..31 (matches B fragment k order).
        const int cb = (lane < 16) ? 0 : 16;
        union { v16h v; v8h h[2]; } a0f, a1f;
        {
            const int ra0 = (r0 + l15) * AP;
            const int ra1 = (r0 + 16 + l15) * AP;
            a0f.h[0] = *(const v8h*)&As[buf][ra0 + cb];
            a0f.h[1] = *(const v8h*)&As[buf][ra0 + cb + 8];
            a1f.h[0] = *(const v8h*)&As[buf][ra1 + cb];
            a1f.h[1] = *(const v8h*)&As[buf][ra1 + cb + 8];
        }
        // ---- B fragments (32x16 f16): lane n holds col N=n, lanes 0-15
        //      K=0..15, lanes 16-31 K=16..31 -> one contiguous 32B load each.
        const int kb = (lane < 16) ? 0 : 16;
        v16h b0 = *(const v16h*)&Bs[buf][(c0 + l15) * BP + kb];
        v16h b1 = *(const v16h*)&Bs[buf][(c0 + 16 + l15) * BP + kb];

        acc00 = __builtin_amdgcn_wmma_f32_16x16x32_f16(false, a0f.v, false, b0,
                                                       (short)0, acc00, false, false);
        acc01 = __builtin_amdgcn_wmma_f32_16x16x32_f16(false, a0f.v, false, b1,
                                                       (short)0, acc01, false, false);
        acc10 = __builtin_amdgcn_wmma_f32_16x16x32_f16(false, a1f.v, false, b0,
                                                       (short)0, acc10, false, false);
        acc11 = __builtin_amdgcn_wmma_f32_16x16x32_f16(false, a1f.v, false, b1,
                                                       (short)0, acc11, false, false);

        if (have_next) store_tile(buf ^ 1);
        __syncthreads();
        buf ^= 1;
    }

    // ---- epilogue: C layout -> lanes 0-15: M=r, lanes 16-31: M=8+r; N=lane&15
    const int colA = tile_n0 + c0 + l15;
    const int colB = colA + 16;
    const int rb0  = tile_m0 + r0 + ((lane < 16) ? 0 : 8);
    const int rb1  = rb0 + 16;

    auto epi = [&](const v8f& acc, int rbase, int col) {
        if (col >= N) return;
        #pragma unroll
        for (int r = 0; r < 8; ++r) {
            const int row = rbase + r;
            if (row >= M) continue;
            const long long base = (long long)row * ldc;
            float v = acc[r] * alpha;
            if (flags & GF_BIAS)  v += bias[col];
            if (flags & GF_RESID) v += resid[base + col];
            if (flags & GF_RELU)  v = fmaxf(v, 0.0f);
            C[base + col] = v;
        }
    };
    epi(acc00, rb0, colA);
    epi(acc01, rb0, colB);
    epi(acc10, rb1, colA);
    epi(acc11, rb1, colB);
}

// ---------------------------------------------------------------------------
// Row softmax over fixed length 512 (one block per row, 2 elems/thread)
// ---------------------------------------------------------------------------
__global__ __launch_bounds__(256) void softmax512_kernel(float* __restrict__ x)
{
    __shared__ float sm[256];
    float* p = x + (long long)blockIdx.x * 512;
    const int t = threadIdx.x;
    float a = p[t];
    float b = p[t + 256];
    sm[t] = fmaxf(a, b);
    __syncthreads();
    for (int s = 128; s > 0; s >>= 1) {
        if (t < s) sm[t] = fmaxf(sm[t], sm[t + s]);
        __syncthreads();
    }
    const float mx = sm[0];
    __syncthreads();
    const float ea = __expf(a - mx);
    const float eb = __expf(b - mx);
    sm[t] = ea + eb;
    __syncthreads();
    for (int s = 128; s > 0; s >>= 1) {
        if (t < s) sm[t] += sm[t + s];
        __syncthreads();
    }
    const float inv = 1.0f / sm[0];
    p[t]       = ea * inv;
    p[t + 256] = eb * inv;
}

// ---------------------------------------------------------------------------
// LayerNorm over fixed length 512 (one block per row)
// ---------------------------------------------------------------------------
__global__ __launch_bounds__(256) void layernorm512_kernel(
    const float* __restrict__ in, const float* __restrict__ g,
    const float* __restrict__ b, float* __restrict__ out)
{
    __shared__ float sm[256];
    const long long base = (long long)blockIdx.x * 512;
    const int t = threadIdx.x;
    float a = in[base + t];
    float c = in[base + t + 256];
    sm[t] = a + c;
    __syncthreads();
    for (int s = 128; s > 0; s >>= 1) { if (t < s) sm[t] += sm[t + s]; __syncthreads(); }
    const float mean = sm[0] * (1.0f / 512.0f);
    __syncthreads();
    const float da = a - mean, dc = c - mean;
    sm[t] = da * da + dc * dc;
    __syncthreads();
    for (int s = 128; s > 0; s >>= 1) { if (t < s) sm[t] += sm[t + s]; __syncthreads(); }
    const float inv = rsqrtf(sm[0] * (1.0f / 512.0f) + 1e-5f);
    out[base + t]       = da * inv * g[t]       + b[t];
    out[base + t + 256] = dc * inv * g[t + 256] + b[t + 256];
}

// ---------------------------------------------------------------------------
// Embedding gather * sqrt(D) + positional encoding (one block per token)
// ---------------------------------------------------------------------------
__global__ __launch_bounds__(256) void embed_pe_kernel(
    const int* __restrict__ tok, const float* __restrict__ emb,
    const float* __restrict__ pe, float* __restrict__ out, int seqlen)
{
    const long long row = blockIdx.x;              // b*seqlen + s
    const int s = (int)(row % seqlen);
    const int t = threadIdx.x;
    const int id = tok[row];
    const float scale = 22.62741699796952f;        // sqrt(512)
    const long long eb = (long long)id * 512;
    const long long pb = (long long)s * 512;
    const long long ob = row * 512;
    out[ob + t]       = emb[eb + t]       * scale + pe[pb + t];
    out[ob + t + 256] = emb[eb + t + 256] * scale + pe[pb + t + 256];
}

// ---------------------------------------------------------------------------
// Host-side orchestration
// ---------------------------------------------------------------------------
static inline void launch_gemm(hipStream_t st, const float* A, const float* B,
    const float* bias, const float* resid, float* C,
    int M, int N, int K, int lda, int ldb, int ldc,
    int batches, long long sAb, long long sAh, long long sBb, long long sBh,
    long long sCb, long long sCh, int heads, float alpha, int flags)
{
    dim3 grid((N + BN - 1) / BN, (M + BM - 1) / BM, batches);
    wmma_gemm_kernel<<<grid, 256, 0, st>>>(A, B, bias, resid, C, M, N, K,
        lda, ldb, ldc, sAb, sAh, sBb, sBh, sCb, sCh, heads, alpha, flags);
}

// Multi-head attention, Sq = Skv = 512, B=8, H=8, dh=64.
static void run_mha(hipStream_t st, const float* xq, const float* xkv,
    const float* wq, const float* bq, const float* wk, const float* bk,
    const float* wv, const float* bv, const float* wo, const float* bo,
    float* Q, float* Kb, float* Vb, float* T3, float* SC, float* outp)
{
    const int R = 8 * 512;                         // B*S rows
    launch_gemm(st, xq,  wq, bq, nullptr, Q,  R, 512, 512, 512, 512, 512,
                1, 0,0,0,0,0,0, 1, 1.0f, GF_BIAS);
    launch_gemm(st, xkv, wk, bk, nullptr, Kb, R, 512, 512, 512, 512, 512,
                1, 0,0,0,0,0,0, 1, 1.0f, GF_BIAS);
    launch_gemm(st, xkv, wv, bv, nullptr, Vb, R, 512, 512, 512, 512, 512,
                1, 0,0,0,0,0,0, 1, 1.0f, GF_BIAS);
    // scores[b,h] = Q_bh @ K_bh^T / sqrt(64)   (64 batched 512x512x64, NT)
    launch_gemm(st, Q, Kb, nullptr, nullptr, SC, 512, 512, 64, 512, 512, 512,
                64, 512LL*512, 64, 512LL*512, 64,
                8LL*512*512, 512LL*512, 8, 0.125f, GF_TRANSB);
    softmax512_kernel<<<8 * 8 * 512, 256, 0, st>>>(SC);
    // T3[b,:,h*64:(h+1)*64] = attn_bh @ V_bh   (64 batched 512x64x512, NN)
    launch_gemm(st, SC, Vb, nullptr, nullptr, T3, 512, 64, 512, 512, 512, 512,
                64, 8LL*512*512, 512LL*512, 512LL*512, 64,
                512LL*512, 64, 8, 1.0f, 0);
    // outp = T3 @ wo + bo + xq (residual fused)
    launch_gemm(st, T3, wo, bo, xq, outp, R, 512, 512, 512, 512, 512,
                1, 0,0,0,0,0,0, 1, 1.0f, GF_BIAS | GF_RESID);
}

// FFN + residual + LayerNorm, in/out in x.
static void run_ffn(hipStream_t st, float* x, const float* w1, const float* b1,
                    const float* w2, const float* b2, float* hid, float* tmp,
                    const float* lng, const float* lnb)
{
    const int R = 8 * 512;
    launch_gemm(st, x, w1, b1, nullptr, hid, R, 2048, 512, 512, 2048, 2048,
                1, 0,0,0,0,0,0, 1, 1.0f, GF_BIAS | GF_RELU);
    launch_gemm(st, hid, w2, b2, x, tmp, R, 512, 2048, 2048, 512, 512,
                1, 0,0,0,0,0,0, 1, 1.0f, GF_BIAS | GF_RESID);
    layernorm512_kernel<<<R, 256, 0, st>>>(tmp, lng, lnb, x);
}

extern "C" void kernel_launch(void* const* d_in, const int* in_sizes, int n_in,
                              void* d_out, int out_size, void* d_ws, size_t ws_size,
                              hipStream_t stream)
{
    (void)in_sizes; (void)n_in; (void)out_size; (void)ws_size;

    // -------- inputs in setup_inputs() dict (insertion) order --------
    const int*   src     = (const int*)  d_in[0];
    const int*   tgt     = (const int*)  d_in[1];
    const float* src_emb = (const float*)d_in[2];
    const float* tgt_emb = (const float*)d_in[3];
    const float* pe      = (const float*)d_in[4];
    // encoder: wq,bq,wk,bk,wv,bv,wo,bo,w1,b1,w2,b2,ln1g,ln1b,ln2g,ln2b
    const float* e_wq  = (const float*)d_in[5];
    const float* e_bq  = (const float*)d_in[6];
    const float* e_wk  = (const float*)d_in[7];
    const float* e_bk  = (const float*)d_in[8];
    const float* e_wv  = (const float*)d_in[9];
    const float* e_bv  = (const float*)d_in[10];
    const float* e_wo  = (const float*)d_in[11];
    const float* e_bo  = (const float*)d_in[12];
    const float* e_w1  = (const float*)d_in[13];
    const float* e_b1  = (const float*)d_in[14];
    const float* e_w2  = (const float*)d_in[15];
    const float* e_b2  = (const float*)d_in[16];
    const float* e_l1g = (const float*)d_in[17];
    const float* e_l1b = (const float*)d_in[18];
    const float* e_l2g = (const float*)d_in[19];
    const float* e_l2b = (const float*)d_in[20];
    // decoder self-attn
    const float* s_wq = (const float*)d_in[21];
    const float* s_bq = (const float*)d_in[22];
    const float* s_wk = (const float*)d_in[23];
    const float* s_bk = (const float*)d_in[24];
    const float* s_wv = (const float*)d_in[25];
    const float* s_bv = (const float*)d_in[26];
    const float* s_wo = (const float*)d_in[27];
    const float* s_bo = (const float*)d_in[28];
    // decoder cross-attn
    const float* c_wq = (const float*)d_in[29];
    const float* c_bq = (const float*)d_in[30];
    const float* c_wk = (const float*)d_in[31];
    const float* c_bk = (const float*)d_in[32];
    const float* c_wv = (const float*)d_in[33];
    const float* c_bv = (const float*)d_in[34];
    const float* c_wo = (const float*)d_in[35];
    const float* c_bo = (const float*)d_in[36];
    // decoder ffn + norms
    const float* d_w1  = (const float*)d_in[37];
    const float* d_b1  = (const float*)d_in[38];
    const float* d_w2  = (const float*)d_in[39];
    const float* d_b2  = (const float*)d_in[40];
    const float* d_l1g = (const float*)d_in[41];
    const float* d_l1b = (const float*)d_in[42];
    const float* d_l2g = (const float*)d_in[43];
    const float* d_l2b = (const float*)d_in[44];
    const float* d_l3g = (const float*)d_in[45];
    const float* d_l3b = (const float*)d_in[46];
    const float* gen_w = (const float*)d_in[47];
    const float* gen_b = (const float*)d_in[48];

    // -------- workspace layout (floats): 7 x 8MB activation bufs + 64MB scores
    const long long NXD = 8LL * 512 * 512;         // B*S*D elements
    float* X  = (float*)d_ws;                      // encoder activations
    float* Y  = X  + NXD;                          // decoder activations
    float* Q  = Y  + NXD;
    float* Kb = Q  + NXD;
    float* Vb = Kb + NXD;
    float* T3 = Vb + NXD;
    float* T4 = T3 + NXD;
    float* SC = T4 + NXD;                          // B*H*S*S (also FFN hidden)

    const int R = 8 * 512;

    // -------- embeddings + positional encoding
    embed_pe_kernel<<<R, 256, 0, stream>>>(src, src_emb, pe, X, 512);
    embed_pe_kernel<<<R, 256, 0, stream>>>(tgt, tgt_emb, pe, Y, 512);

    // -------- encoder layers
    for (int i = 0; i < 3; ++i) {
        const long long oDD = (long long)i * 512 * 512;
        const long long oD  = (long long)i * 512;
        const long long oDF = (long long)i * 512 * 2048;
        const long long oF  = (long long)i * 2048;
        run_mha(stream, X, X,
                e_wq + oDD, e_bq + oD, e_wk + oDD, e_bk + oD,
                e_wv + oDD, e_bv + oD, e_wo + oDD, e_bo + oD,
                Q, Kb, Vb, T3, SC, T4);
        layernorm512_kernel<<<R, 256, 0, stream>>>(T4, e_l1g + oD, e_l1b + oD, X);
        run_ffn(stream, X, e_w1 + oDF, e_b1 + oF, e_w2 + oDF, e_b2 + oD,
                SC, T4, e_l2g + oD, e_l2b + oD);
    }

    // -------- decoder layers
    for (int i = 0; i < 3; ++i) {
        const long long oDD = (long long)i * 512 * 512;
        const long long oD  = (long long)i * 512;
        const long long oDF = (long long)i * 512 * 2048;
        const long long oF  = (long long)i * 2048;
        // self-attention
        run_mha(stream, Y, Y,
                s_wq + oDD, s_bq + oD, s_wk + oDD, s_bk + oD,
                s_wv + oDD, s_bv + oD, s_wo + oDD, s_bo + oD,
                Q, Kb, Vb, T3, SC, T4);
        layernorm512_kernel<<<R, 256, 0, stream>>>(T4, d_l1g + oD, d_l1b + oD, Y);
        // cross-attention (K/V from encoder output X)
        run_mha(stream, Y, X,
                c_wq + oDD, c_bq + oD, c_wk + oDD, c_bk + oD,
                c_wv + oDD, c_bv + oD, c_wo + oDD, c_bo + oD,
                Q, Kb, Vb, T3, SC, T4);
        layernorm512_kernel<<<R, 256, 0, stream>>>(T4, d_l2g + oD, d_l2b + oD, Y);
        // FFN
        run_ffn(stream, Y, d_w1 + oDF, d_b1 + oF, d_w2 + oDF, d_b2 + oD,
                SC, T4, d_l3g + oD, d_l3b + oD);
    }

    // -------- generator: logits = Y @ gen_w + gen_b  (4096 x 10000 x 512)
    launch_gemm(stream, Y, gen_w, gen_b, nullptr, (float*)d_out,
                R, 10000, 512, 512, 10000, 10000,
                1, 0,0,0,0,0,0, 1, 1.0f, GF_BIAS);
}
